// HardOrLayer_51342039056576
// MI455X (gfx1250) — 8-bit matrix operand order repaired and checked by reading, compiled, not run on hardware
//
#include <hip/hip_runtime.h>
#include <stdint.h>


constexpr int kBatch   = 4096;
constexpr int kInDim   = 8192;
constexpr int kLayer   = 8192;

constexpr int BM = 128;
constexpr int BN = 128;
constexpr int BK = 128;
constexpr int LDK = 144;
constexpr int kThreads = 256;
constexpr int kBufBytes = (BM + BN) * LDK;

typedef __attribute__((ext_vector_type(16))) int   v16i;
typedef __attribute__((ext_vector_type(8)))  float v8f;
typedef __attribute__((ext_vector_type(4)))  int   v4i;

#if defined(__AMDGCN__) && \
    __has_builtin(__builtin_amdgcn_global_load_async_to_lds_b128) && \
    __has_builtin(__builtin_amdgcn_s_wait_asynccnt)
#define USE_ASYNC_LDS 1
#else
#define USE_ASYNC_LDS 0
#endif

__device__ __forceinline__ void cp16_g2l(const unsigned char* g, unsigned char* l) {
#if USE_ASYNC_LDS
  __builtin_amdgcn_global_load_async_to_lds_b128(
      (__attribute__((address_space(1))) v4i*)g,
      (__attribute__((address_space(3))) v4i*)l,
      0, 0);
#else
  *(uint4*)l = *(const uint4*)g;
#endif
}

__device__ __forceinline__ void wait_fills() {
#if USE_ASYNC_LDS
  __builtin_amdgcn_s_wait_asynccnt(0);
#endif
}

__global__ __launch_bounds__(kThreads) void hard_or_wmma_fp8_kernel(
    const unsigned char* __restrict__ x,
    const unsigned char* __restrict__ w,
    float* __restrict__ out) {
  extern __shared__ unsigned char smem[];

  const int tid  = threadIdx.x;
  const int wave = tid >> 5;
  const int lane = tid & 31;
  const int half = lane >> 4;
  const int r    = lane & 15;

  const int m0 = blockIdx.y * BM;
  const int n0 = blockIdx.x * BN;
  const unsigned char* gA = x + (size_t)m0 * kInDim;
  const unsigned char* gB = w + (size_t)n0 * kInDim;

  auto fill = [&](int buf, int k) {
    unsigned char* lA = smem + (size_t)buf * kBufBytes;
    unsigned char* lB = lA + (size_t)BM * LDK;
#pragma unroll
    for (int j = 0; j < 4; ++j) {
      const int row = j * 32 + (tid >> 3);
      const int col = (tid & 7) * 16;
      cp16_g2l(gA + (size_t)row * kInDim + k + col, lA + row * LDK + col);
      uint4 d = *(const uint4*)(gB + (size_t)row * kInDim + k + col);
      d.x *= 0x38u; d.y *= 0x38u; d.z *= 0x38u; d.w *= 0x38u;
      *(uint4*)(lB + row * LDK + col) = d;
    }
  };

  auto load_a = [&](const unsigned char* tile) -> v16i {
    const unsigned char* row = tile + (size_t)(16 * wave + r) * LDK + 16 * half;
    v16i f;
#pragma unroll
    for (int p = 0; p < 8; ++p) {
      const unsigned long long d = *(const unsigned long long*)(row + 32 * (p >> 1) + 8 * (p & 1));
      f[2 * p + 0] = (int)((unsigned)d         * 0x38u);
      f[2 * p + 1] = (int)((unsigned)(d >> 32) * 0x38u);
    }
    return f;
  };

  auto load_b = [&](const unsigned char* tile, int nsub) -> v16i {
    const unsigned char* row = tile + (size_t)(16 * nsub + r) * LDK + 16 * half;
    v16i f;
#pragma unroll
    for (int g = 0; g < 4; ++g) {
      const v4i d = *(const v4i*)(row + 32 * g);
      f[4 * g + 0] = d.x;
      f[4 * g + 1] = d.y;
      f[4 * g + 2] = d.z;
      f[4 * g + 3] = d.w;
    }
    return f;
  };

  v8f acc[8] = {};

  fill(0, 0);
  wait_fills();
  __syncthreads();

  constexpr int kSteps = kInDim / BK;
  for (int kt = 0; kt < kSteps; ++kt) {
    const int cur = kt & 1;
    if (kt + 1 < kSteps) fill(cur ^ 1, (kt + 1) * BK);

    const unsigned char* tA = smem + (size_t)cur * kBufBytes;
    const unsigned char* tB = tA + (size_t)BM * LDK;

    const v16i a = load_a(tA);
#pragma unroll
    for (int n = 0; n < 8; ++n) {
      const v16i b = load_b(tB, n);
      acc[n] = __builtin_amdgcn_wmma_f32_16x16x128_fp8_fp8(
          a, b, (short)0, acc[n], false, false);
    }

    wait_fills();
    __syncthreads();
  }

  float* orow = out + ((size_t)(m0 + 16 * wave + 8 * half)) * kLayer + n0 + r;
#pragma unroll
  for (int n = 0; n < 8; ++n) {
#pragma unroll
    for (int rr = 0; rr < 8; ++rr) {
      orow[(size_t)rr * kLayer + 16 * n] = (acc[n][rr] > 0.0f) ? 1.0f : 0.0f;
    }
  }
}

extern "C" void kernel_launch(void* const* d_in, const int* in_sizes, int n_in,
                              void* d_out, int out_size, void* d_ws, size_t ws_size,
                              hipStream_t stream) {
  (void)in_sizes; (void)n_in; (void)out_size; (void)d_ws; (void)ws_size;
  const unsigned char* x = (const unsigned char*)d_in[0];
  const unsigned char* w = (const unsigned char*)d_in[1];
  float* out = (float*)d_out;

  dim3 grid(kLayer / BN, kBatch / BM);
  dim3 block(kThreads);
  const size_t shmem = (size_t)2 * kBufBytes;

  hard_or_wmma_fp8_kernel<<<grid, block, shmem, stream>>>(x, w, out);
}
